// FeatureAlign_34677565947928
// MI455X (gfx1250) — matrix operand order repaired and checked by reading, not compiled, not run on hardware
//
#include <hip/hip_runtime.h>
#include <math.h>

// ---------------------------------------------------------------------------
// FeatureAlign for MI455X (gfx1250, wave32, WMMA).
// All convs lowered to GEMM on v_wmma_f32_16x16x32_f16 (f16 in, f32 acc).
// Operands are kept fragment-resident in memory:
//   A: weights pre-packed into per-tile lane-major v16h fragments
//   X: activations stored [n][K] (K contiguous) -> B fragment = one 32B load
// ---------------------------------------------------------------------------

typedef _Float16 f16;
typedef __attribute__((ext_vector_type(16))) _Float16 v16h;
typedef __attribute__((ext_vector_type(8)))  float    v8f;

constexpr int kC     = 128;
constexpr int kH     = 128;
constexpr int kW     = 128;
constexpr int kHW    = kH * kW;          // 16384
constexpr int kN     = 2 * kHW;          // 32768 pixels (n = b*HW + y*W + x)
constexpr int kDG    = 8;
constexpr int kTaps  = 9;
constexpr int kKom   = kTaps * kC;       // 1152, k = tap*128 + ci
constexpr int kMom   = 3 * kDG * kTaps;  // 216
constexpr int kMomP  = 256;              // padded M for om conv

// A-fragment k-mapping for 16-bit A 16x32, wave32: contiguous k = 16*(lane/16) + h.
__device__ __forceinline__ int a_kmap(int h, int lane) {
  return h + ((lane & 16) ? 16 : 0);
}

// ---------------------------------------------------------------------------
// GEMM: Y[M][N] (f32) = A[M][K] (f16, tiled) * X[n][K] (f16) (+bias)
// 256 threads = 8 waves; wave wv owns rows m0+16wv..+16; NT=4 n-tiles (16x64).
// A tiles flat: ((mt*Kt + kt)*32 + lane)*16 + h
// ---------------------------------------------------------------------------
template <int K, int NT>
__global__ __launch_bounds__(256)
void gemm_tiles(const f16* __restrict__ A, const f16* __restrict__ X,
                float* __restrict__ Y, const float* __restrict__ bias, int N) {
  constexpr int Kt = K >> 5;
  const int lane = threadIdx.x & 31;
  const int wv   = threadIdx.x >> 5;
  const int n0   = blockIdx.x * (16 * NT);
  const int m0   = blockIdx.y * 128;
  const int mt   = (m0 >> 4) + wv;
  const int bn   = lane & 15;
  const int bk   = lane & 16;          // 0 or 16 -> K offset of this lane
  const f16* Ap  = A + ((size_t)mt * Kt * 32 + lane) * 16;
  const f16* Xp  = X + (size_t)(n0 + bn) * K + bk;
  v8f acc[NT] = {};
#pragma unroll 4
  for (int ks = 0; ks < Kt; ++ks) {
    const v16h a = *(const v16h*)(Ap + ks * 512);
    __builtin_prefetch(Ap + (ks + 1) * 512, 0, 1);           // global_prefetch_b8
    __builtin_prefetch(Xp + (size_t)(ks + 1) * 32, 0, 1);
#pragma unroll
    for (int t = 0; t < NT; ++t) {
      const v16h b = *(const v16h*)(Xp + (size_t)t * 16 * K + ks * 32);
      acc[t] = __builtin_amdgcn_wmma_f32_16x16x32_f16(false, a, false, b,
                                                      (short)0, acc[t], false, false);
    }
  }
#pragma unroll
  for (int t = 0; t < NT; ++t) {
    const int n = n0 + t * 16 + bn;
#pragma unroll
    for (int r = 0; r < 8; ++r) {
      const int m = m0 + wv * 16 + r + ((lane & 16) ? 8 : 0);
      float v = acc[t][r];
      if (bias) v += bias[m];
      Y[(size_t)m * N + n] = v;
    }
  }
}

// ---------------------------------------------------------------------------
// 3x3 conv (pad 1) implicit GEMM; X = [n][128] f16. Each 32-wide K-step is one
// tap & 16 contiguous ci per lane -> B fragment is one conditional v16h load.
// ---------------------------------------------------------------------------
template <int NT>
__global__ __launch_bounds__(256)
void conv3x3_tiles(const f16* __restrict__ A, const f16* __restrict__ X,
                   float* __restrict__ Y, int N) {
  constexpr int Kt = kKom >> 5;   // 36
  const int lane = threadIdx.x & 31;
  const int wv   = threadIdx.x >> 5;
  const int n0   = blockIdx.x * (16 * NT);
  const int m0   = blockIdx.y * 128;
  const int mt   = (m0 >> 4) + wv;
  const int bn   = lane & 15;
  const int bk   = lane & 16;
  const f16* Ap  = A + ((size_t)mt * Kt * 32 + lane) * 16;
  const v16h zero = {};
  v8f acc[NT] = {};
#pragma unroll 4
  for (int ks = 0; ks < Kt; ++ks) {
    const v16h a = *(const v16h*)(Ap + ks * 512);
    const int kbase = ks * 32 + bk;       // tap*128 + ci0 for this lane
    const int tap = kbase >> 7;
    const int ci0 = kbase & 127;
    const int dy  = tap / 3 - 1;
    const int dx  = tap % 3 - 1;
#pragma unroll
    for (int t = 0; t < NT; ++t) {
      const int n  = n0 + t * 16 + bn;
      const int b  = n >> 14;
      const int hw = n & (kHW - 1);
      const int yy = (hw >> 7) + dy;
      const int xx = (hw & 127) + dx;
      v16h bf = zero;
      if (yy >= 0 && yy < kH && xx >= 0 && xx < kW)
        bf = *(const v16h*)(X + ((size_t)((b << 14) + yy * kW + xx)) * kC + ci0);
      acc[t] = __builtin_amdgcn_wmma_f32_16x16x32_f16(false, a, false, bf,
                                                      (short)0, acc[t], false, false);
    }
  }
#pragma unroll
  for (int t = 0; t < NT; ++t) {
    const int n = n0 + t * 16 + bn;
#pragma unroll
    for (int r = 0; r < 8; ++r) {
      const int m = m0 + wv * 16 + r + ((lane & 16) ? 8 : 0);
      Y[(size_t)m * N + n] = acc[t][r];
    }
  }
}

// ---------------------------------------------------------------------------
// Weight packing into fragment-ready tiles
// ---------------------------------------------------------------------------
// 1x1 weights [M][K] fp32 -> tiled f16
__global__ void pack_a_tiles_k(const float* __restrict__ w, f16* __restrict__ A,
                               int M, int K) {
  const int total = M * K;
  int idx = blockIdx.x * 256 + threadIdx.x;
  if (idx >= total) return;
  const int h    = idx & 15;
  const int lane = (idx >> 4) & 31;
  const int tile = idx >> 9;
  const int Kt   = K >> 5;
  const int mt   = tile / Kt, kt = tile - mt * Kt;
  const int m = mt * 16 + (lane & 15);
  const int k = kt * 32 + a_kmap(h, lane);
  A[idx] = (f16)w[(size_t)m * K + k];
}

// 3x3 weights [Mout][128][3][3] fp32 -> tiled f16 with k = tap*128+ci, M padded
__global__ void pack_a3_tiles_k(const float* __restrict__ w, f16* __restrict__ A,
                                int Mout, int Mp) {
  const int total = Mp * kKom;
  int idx = blockIdx.x * 256 + threadIdx.x;
  if (idx >= total) return;
  const int h    = idx & 15;
  const int lane = (idx >> 4) & 31;
  const int tile = idx >> 9;
  constexpr int Kt = kKom >> 5;
  const int mt = tile / Kt, kt = tile - mt * Kt;
  const int m = mt * 16 + (lane & 15);
  const int k = kt * 32 + a_kmap(h, lane);
  const int t = k >> 7, ci = k & 127;
  A[idx] = (m < Mout) ? (f16)w[((size_t)m * kC + ci) * 9 + t] : (f16)0.f;
}

// ---------------------------------------------------------------------------
// Elementwise / reduction kernels
// ---------------------------------------------------------------------------
// NCHW fp32 -> [n][128] f16 (K-contiguous writes)
__global__ void pack_nchw_k(const float* __restrict__ s, f16* __restrict__ d) {
  int idx = blockIdx.x * 256 + threadIdx.x;          // n*128 + c
  if (idx >= kC * kN) return;
  const int c = idx & 127, n = idx >> 7;
  const int b = n >> 14, hw = n & (kHW - 1);
  d[idx] = (f16)s[((size_t)(b * kC + c) << 14) + hw];
}

// bilinear 2x upsample (half-pixel, edge clamp): Up fp32 [c][n] planes for the
// sampler + f16 2*up into dst[n*ld + 128 + c] (second half of Xoff)
__global__ void upsample_k(const float* __restrict__ fs, float* __restrict__ up,
                           f16* __restrict__ dst, int ld) {
  int idx = blockIdx.x * 256 + threadIdx.x;          // c*N + n
  if (idx >= kC * kN) return;
  const int c = idx >> 15, n = idx & (kN - 1);
  const int b = n >> 14, hw = n & (kHW - 1);
  const int y = hw >> 7, x = hw & 127;
  const float sy = 0.5f * y - 0.25f, sx = 0.5f * x - 0.25f;
  const int y0 = (int)floorf(sy), x0 = (int)floorf(sx);
  const float wy = sy - y0, wx = sx - x0;
  const int y0c = min(max(y0, 0), 63), y1c = min(max(y0 + 1, 0), 63);
  const int x0c = min(max(x0, 0), 63), x1c = min(max(x0 + 1, 0), 63);
  const float* p = fs + ((size_t)(b * kC + c) << 12);
  const float v = (1.f - wy) * ((1.f - wx) * p[y0c * 64 + x0c] + wx * p[y0c * 64 + x1c])
                +        wy  * ((1.f - wx) * p[y1c * 64 + x0c] + wx * p[y1c * 64 + x1c]);
  up[idx] = v;
  dst[(size_t)n * ld + 128 + c] = (f16)(2.f * v);
}

// per-channel sum / sumsq over N (training-mode BN stats); Y is [M][N]
__global__ __launch_bounds__(256)
void channel_stats_k(const float* __restrict__ Y, float* __restrict__ stats, int N) {
  __shared__ float ssum[256], ssq[256];
  const int c = blockIdx.x;
  const float* row = Y + (size_t)c * N;
  float s = 0.f, q = 0.f;
  for (int i = threadIdx.x; i < N; i += 256) { float v = row[i]; s += v; q += v * v; }
  ssum[threadIdx.x] = s; ssq[threadIdx.x] = q;
  __syncthreads();
  for (int o = 128; o > 0; o >>= 1) {
    if (threadIdx.x < o) { ssum[threadIdx.x] += ssum[threadIdx.x + o];
                           ssq[threadIdx.x]  += ssq[threadIdx.x + o]; }
    __syncthreads();
  }
  if (threadIdx.x == 0) { stats[2 * c] = ssum[0]; stats[2 * c + 1] = ssq[0]; }
}

// BN (training) + ReLU, pack f16 into [n][ld] (up to two destinations)
__global__ void bn_relu_pack_k(const float* __restrict__ Y, const float* __restrict__ stats,
                               const float* __restrict__ g, const float* __restrict__ bt,
                               f16* __restrict__ d1, int ld1,
                               f16* __restrict__ d2, int ld2, int N) {
  int idx = blockIdx.x * 256 + threadIdx.x;          // n*128 + c
  if (idx >= kC * N) return;
  const int c = idx & 127, n = idx >> 7;
  const float mean = stats[2 * c] / (float)N;
  const float var  = stats[2 * c + 1] / (float)N - mean * mean;
  const float sc = g[c] * rsqrtf(var + 1e-5f);
  const float sh = bt[c] - mean * sc;
  const float v  = fmaxf(Y[(size_t)c * N + n] * sc + sh, 0.f);
  d1[(size_t)n * ld1 + c] = (f16)v;
  if (d2) d2[(size_t)n * ld2 + c] = (f16)v;
}

// ReLU + pack f16 into dst[n*ld + coloff + c]
__global__ void relu_pack_k(const float* __restrict__ Y, f16* __restrict__ d,
                            int ld, int coloff, int N) {
  int idx = blockIdx.x * 256 + threadIdx.x;          // n*128 + c
  if (idx >= kC * N) return;
  const int c = idx & 127, n = idx >> 7;
  d[(size_t)n * ld + coloff + c] = (f16)fmaxf(Y[(size_t)c * N + n], 0.f);
}

// Final BN+ReLU, fp32 NCHW output
__global__ void bn_relu_out_k(const float* __restrict__ Y, const float* __restrict__ stats,
                              const float* __restrict__ g, const float* __restrict__ bt,
                              float* __restrict__ out, int N) {
  int idx = blockIdx.x * 256 + threadIdx.x;          // c*N + n
  if (idx >= kC * N) return;
  const int c = idx / N, n = idx - c * N;
  const int b = n >> 14, hw = n & (kHW - 1);
  const float mean = stats[2 * c] / (float)N;
  const float var  = stats[2 * c + 1] / (float)N - mean * mean;
  const float sc = g[c] * rsqrtf(var + 1e-5f);
  const float sh = bt[c] - mean * sc;
  out[((size_t)(b * kC + c) << 14) + hw] = fmaxf(Y[idx] * sc + sh, 0.f);
}

// ---------------------------------------------------------------------------
// Modulated deformable sampling -> im2col cols[n][1152] f16 (k = tap*128+ci).
// om: Y buffer rows [0..215][N] fp32 (dy/dx interleaved 0..143, mask 144..215).
// One thread per (tap, dg, n) -> writes 16 contiguous halves.
// ---------------------------------------------------------------------------
__global__ void build_dcn_cols_k(const float* __restrict__ om, const float* __restrict__ b_om,
                                 const float* __restrict__ up, f16* __restrict__ cols, int N) {
  int idx = blockIdx.x * 256 + threadIdx.x;
  if (idx >= kTaps * kDG * kN) return;
  const int n  = idx & (kN - 1);
  const int dg = (idx >> 15) & 7;
  const int t  = idx >> 18;
  const int oc = dg * kTaps + t;                       // 0..71
  const float dyo = om[(size_t)(2 * oc)     * N + n] + b_om[2 * oc];
  const float dxo = om[(size_t)(2 * oc + 1) * N + n] + b_om[2 * oc + 1];
  const float ml  = om[(size_t)(144 + oc)   * N + n] + b_om[144 + oc];
  const float mk  = 1.f / (1.f + expf(-ml));
  const int b = n >> 14;
  const int hw = n & (kHW - 1);
  const float py = (float)(hw >> 7)  + (float)(t / 3 - 1) + dyo;
  const float px = (float)(hw & 127) + (float)(t % 3 - 1) + dxo;
  const float fy = floorf(py), fx = floorf(px);
  const int y0 = (int)fy, x0 = (int)fx;
  const float wy1 = py - fy, wx1 = px - fx;
  const float vy0 = (y0 >= 0 && y0 < kH) ? 1.f : 0.f;
  const float vy1 = (y0 + 1 >= 0 && y0 + 1 < kH) ? 1.f : 0.f;
  const float vx0 = (x0 >= 0 && x0 < kW) ? 1.f : 0.f;
  const float vx1 = (x0 + 1 >= 0 && x0 + 1 < kW) ? 1.f : 0.f;
  const float a00 = (1.f - wy1) * (1.f - wx1) * vy0 * vx0 * mk;
  const float a01 = (1.f - wy1) * wx1         * vy0 * vx1 * mk;
  const float a10 = wy1 * (1.f - wx1)         * vy1 * vx0 * mk;
  const float a11 = wy1 * wx1                 * vy1 * vx1 * mk;
  const int cy0 = min(max(y0, 0), kH - 1), cy1 = min(max(y0 + 1, 0), kH - 1);
  const int cx0 = min(max(x0, 0), kW - 1), cx1 = min(max(x0 + 1, 0), kW - 1);
  const int base = b << 14;
  const int i00 = base + cy0 * kW + cx0, i01 = base + cy0 * kW + cx1;
  const int i10 = base + cy1 * kW + cx0, i11 = base + cy1 * kW + cx1;
  f16* dst = cols + (size_t)n * kKom + t * kC + dg * 16;
#pragma unroll
  for (int ci = 0; ci < 16; ++ci) {
    const float* pl = up + (size_t)(dg * 16 + ci) * N;
    dst[ci] = (f16)(a00 * pl[i00] + a01 * pl[i01] + a10 * pl[i10] + a11 * pl[i11]);
  }
}

// ---------------------------------------------------------------------------
extern "C" void kernel_launch(void* const* d_in, const int* in_sizes, int n_in,
                              void* d_out, int out_size, void* d_ws, size_t ws_size,
                              hipStream_t stream) {
  const float* feat_l = (const float*)d_in[0];
  const float* feat_s = (const float*)d_in[1];
  const float* w_fsm  = (const float*)d_in[2];
  const float* g_fsm  = (const float*)d_in[3];
  const float* b_fsm  = (const float*)d_in[4];
  const float* w_off  = (const float*)d_in[5];
  const float* g_off  = (const float*)d_in[6];
  const float* b_off  = (const float*)d_in[7];
  const float* w_om   = (const float*)d_in[8];
  const float* b_om   = (const float*)d_in[9];
  const float* w_dcn  = (const float*)d_in[10];
  const float* b_dcn  = (const float*)d_in[11];
  const float* w_fus  = (const float*)d_in[12];
  const float* g_fus  = (const float*)d_in[13];
  const float* b_fus  = (const float*)d_in[14];
  (void)in_sizes; (void)n_in; (void)out_size; (void)ws_size;

  char* ws = (char*)d_ws;
  size_t off = 0;
  auto take = [&](size_t bytes) -> char* {
    char* p = ws + off;
    off = (off + bytes + 255) & ~(size_t)255;
    return p;
  };
  f16*   Xl    = (f16*)  take((size_t)kC * kN * 2);          // [n][128]
  f16*   Afsm  = (f16*)  take((size_t)128 * 128 * 2);
  f16*   Aoff  = (f16*)  take((size_t)128 * 256 * 2);
  f16*   Afus  = (f16*)  take((size_t)128 * 256 * 2);
  f16*   Aom   = (f16*)  take((size_t)kMomP * kKom * 2);
  f16*   Adcn  = (f16*)  take((size_t)128 * kKom * 2);
  float* stats = (float*)take(1024);
  float* Yb    = (float*)take((size_t)256 * kN * 4);         // shared GEMM out [M][N]
  f16*   Xoff  = (f16*)  take((size_t)256 * kN * 2);         // [n][256] / later [n][128] Xom
  f16*   Xfus  = (f16*)  take((size_t)256 * kN * 2);         // [n][256]
  float* Up    = (float*)take((size_t)kC * kN * 4);          // feat_up fp32 [c][n]
  f16*   Cols  = (f16*)  take((size_t)kKom * kN * 2);        // [n][1152]

  const dim3 blk(256);
  const int EW = (kC * kN) / 256;                            // 16384 blocks
  const dim3 gemm_grid(kN / 64, 1);

  // --- packs -------------------------------------------------------------
  pack_a_tiles_k<<<(128 * 128 + 255) / 256, blk, 0, stream>>>(w_fsm, Afsm, 128, 128);
  pack_a_tiles_k<<<(128 * 256 + 255) / 256, blk, 0, stream>>>(w_off, Aoff, 128, 256);
  pack_a_tiles_k<<<(128 * 256 + 255) / 256, blk, 0, stream>>>(w_fus, Afus, 128, 256);
  pack_a3_tiles_k<<<(kMomP * kKom + 255) / 256, blk, 0, stream>>>(w_om, Aom, kMom, kMomP);
  pack_a3_tiles_k<<<(128 * kKom + 255) / 256, blk, 0, stream>>>(w_dcn, Adcn, 128, 128);
  pack_nchw_k<<<EW, blk, 0, stream>>>(feat_l, Xl);
  upsample_k<<<EW, blk, 0, stream>>>(feat_s, Up, Xoff, 256);

  // --- fsm: 1x1 conv + BN + ReLU -> feat_arm (cols 0..127 of Xoff & Xfus) -
  gemm_tiles<128, 4><<<gemm_grid, blk, 0, stream>>>(Afsm, Xl, Yb, nullptr, kN);
  channel_stats_k<<<128, blk, 0, stream>>>(Yb, stats, kN);
  bn_relu_pack_k<<<EW, blk, 0, stream>>>(Yb, stats, g_fsm, b_fsm, Xoff, 256, Xfus, 256, kN);

  // --- off: 1x1 conv on concat(arm, 2*up) + BN + ReLU -> off_feat (Xom) ---
  gemm_tiles<256, 4><<<gemm_grid, blk, 0, stream>>>(Aoff, Xoff, Yb, nullptr, kN);
  channel_stats_k<<<128, blk, 0, stream>>>(Yb, stats, kN);
  bn_relu_pack_k<<<EW, blk, 0, stream>>>(Yb, stats, g_off, b_off,
                                         (f16*)Xoff /*Xom [n][128]*/, 128, nullptr, 0, kN);

  // --- om: 3x3 conv (implicit GEMM, M padded to 256); bias added in sampler
  conv3x3_tiles<4><<<dim3(kN / 64, 2), blk, 0, stream>>>(Aom, Xoff, Yb, kN);

  // --- deformable sampling -> cols, then DCN GEMM + bias ------------------
  build_dcn_cols_k<<<(kTaps * kDG * kN) / 256, blk, 0, stream>>>(Yb, b_om, Up, Cols, kN);
  gemm_tiles<1152, 4><<<gemm_grid, blk, 0, stream>>>(Adcn, Cols, Yb, b_dcn, kN);
  relu_pack_k<<<EW, blk, 0, stream>>>(Yb, Xfus, 256, 128, kN);

  // --- fus: 1x1 conv on concat(arm, align) + BN + ReLU -> output ----------
  gemm_tiles<256, 4><<<gemm_grid, blk, 0, stream>>>(Afus, Xfus, Yb, nullptr, kN);
  channel_stats_k<<<128, blk, 0, stream>>>(Yb, stats, kN);
  bn_relu_out_k<<<EW, blk, 0, stream>>>(Yb, stats, g_fus, b_fus, (float*)d_out, kN);
}
